// SparseTransformerEncoder_90838558310637
// MI455X (gfx1250) — matrix operand order repaired and checked by reading, not compiled, not run on hardware
//
#include <hip/hip_runtime.h>
#include <math.h>

// ---------------------------------------------------------------------------
// Sparse transformer encoder forward, CDNA5 (gfx1250, wave32, WMMA bf16,
// async global->LDS staging for GEMM weight tiles).
// ---------------------------------------------------------------------------

#define S_LEN   2048
#define BATCH   4
#define DMODEL  256
#define NHEAD   8
#define HDIM    32
#define MLPD    1024
#define NLAYER  4
#define VECQ    8
#define MBLK    256          // S_LEN / VECQ
#define QKVW    (3 * DMODEL)
#define ROWS    (S_LEN * BATCH)   // 8192 token rows (s-major, b-minor)

typedef __attribute__((ext_vector_type(16))) __bf16 v16bf;
typedef __attribute__((ext_vector_type(8)))  __bf16 v8bf;
typedef __attribute__((ext_vector_type(8)))  float  v8f;

__device__ __forceinline__ v8f vzero8() {
  v8f z;
#pragma unroll
  for (int i = 0; i < 8; i++) z[i] = 0.f;
  return z;
}

// Load 16 bf16 as two 16-byte chunks (chunk1 offset chosen by caller).
__device__ __forceinline__ v16bf ld16(const __bf16* p0, const __bf16* p1) {
  v8bf a = *(const v8bf*)p0;
  v8bf b = *(const v8bf*)p1;
  v16bf r;
#pragma unroll
  for (int i = 0; i < 8; i++) { r[i] = a[i]; r[i + 8] = b[i]; }
  return r;
}

__device__ __forceinline__ v8f wmma_bf16(v16bf a, v16bf b, v8f c) {
  // (neg_a, A, neg_b, B, c_mod, C, reuse_a, reuse_b)
  return __builtin_amdgcn_wmma_f32_16x16x32_bf16(false, a, false, b,
                                                 (short)0, c, false, false);
}

// ---------------------------------------------------------------------------
// f32 -> bf16 cast (weights, once per launch; weights then live in L2 as bf16)
// ---------------------------------------------------------------------------
__global__ void k_cast_bf16(const float* __restrict__ in,
                            __bf16* __restrict__ out, int n) {
  int i = blockIdx.x * blockDim.x + threadIdx.x;
  int stride = gridDim.x * blockDim.x;
  for (; i < n; i += stride) out[i] = (__bf16)in[i];
}

// ---------------------------------------------------------------------------
// Embedding gather * sqrt(D) + sinusoidal positional encoding -> h [S,B,D] f32
// One thread per (s,b,dim-pair).
// ---------------------------------------------------------------------------
__global__ __launch_bounds__(256) void k_embed(const int* __restrict__ x,
                                               const float* __restrict__ emb,
                                               float* __restrict__ h) {
  int idx = blockIdx.x * 256 + threadIdx.x;   // S*B*128 threads
  int d2 = idx & 127;
  int sb = idx >> 7;
  int b  = sb & (BATCH - 1);
  int s  = sb >> 2;
  int tok = x[b * S_LEN + s];
  float e0 = emb[(size_t)tok * DMODEL + 2 * d2];
  float e1 = emb[(size_t)tok * DMODEL + 2 * d2 + 1];
  // div = exp(-(2*d2)*ln(10000)/D)
  float div = __expf((float)(2 * d2) * (-9.210340371976184f / (float)DMODEL));
  float ang = (float)s * div;
  float* hp = h + (size_t)(s * BATCH + b) * DMODEL + 2 * d2;
  hp[0] = e0 * 16.f + __sinf(ang);
  hp[1] = e1 * 16.f + __cosf(ang);
}

// ---------------------------------------------------------------------------
// LayerNorm over D=256: one wave per token row, 8 elems/lane, f32 in -> bf16
// ---------------------------------------------------------------------------
__global__ __launch_bounds__(256) void k_ln(const float* __restrict__ in,
                                            const float* __restrict__ g,
                                            const float* __restrict__ bta,
                                            __bf16* __restrict__ out) {
  const int lane = threadIdx.x & 31;
  const int row  = blockIdx.x * 8 + (threadIdx.x >> 5);
  const float* p = in + (size_t)row * DMODEL;
  float x[8];
  float s = 0.f, s2 = 0.f;
#pragma unroll
  for (int i = 0; i < 8; i++) {
    x[i] = p[lane + 32 * i];
    s += x[i];
    s2 += x[i] * x[i];
  }
#pragma unroll
  for (int m = 16; m >= 1; m >>= 1) {
    s  += __shfl_xor(s,  m, 32);
    s2 += __shfl_xor(s2, m, 32);
  }
  const float mu   = s * (1.f / DMODEL);
  const float var  = s2 * (1.f / DMODEL) - mu * mu;
  const float rstd = rsqrtf(var + 1e-5f);
  __bf16* o = out + (size_t)row * DMODEL;
#pragma unroll
  for (int i = 0; i < 8; i++) {
    int d = lane + 32 * i;
    o[d] = (__bf16)((x[i] - mu) * rstd * g[d] + bta[d]);
  }
}

// ---------------------------------------------------------------------------
// WMMA bf16 GEMM:  C[M,N] = A[M,K] @ W[N,K]^T (+bias) (+resid) -> bf16/f32
// 256 threads = 8 waves; wave computes a 16x64 output strip (4 WMMA N-tiles).
// All 8 waves share the same 64-column W panel, so the 64x32 W chunk is
// staged ONCE per block into LDS with GLOBAL_LOAD_ASYNC_TO_LDS_B128
// (double-buffered, ASYNCcnt-tracked) -> 8x less W vmem traffic, copy
// overlapped with WMMA.  K % 32 == 0, N % 64 == 0, M % 128 == 0.
// QSCALE: multiply columns < qcols by qscale AFTER the bias (q head scaling).
// ---------------------------------------------------------------------------
template <bool WBF, bool WF32, bool RESID, bool QSCALE>
__global__ __launch_bounds__(256) void k_gemm(
    const __bf16* __restrict__ A, const __bf16* __restrict__ W,
    const float* __restrict__ bias,
    __bf16* __restrict__ outb, float* __restrict__ outf,
    const float* __restrict__ resid,
    int M, int N, int K, float qscale, int qcols) {
  __shared__ __bf16 wbuf[2][64 * 32];   // double-buffered 64x32 W chunk (8KB)

  const int tid  = threadIdx.x;
  const int lane = tid & 31;
  const int wave = tid >> 5;
  const int li = lane & 15;
  const int hi = lane >> 4;
  const int n0 = blockIdx.x * 64;
  const int m0 = blockIdx.y * 128 + wave * 16;

  // async-copy assignment: thread -> (row 0..63, 8-elem group 0..3)
  const int wrow = tid >> 2;
  const int wcol = (tid & 3) * 8;
  const __bf16* wsrc0 = W + (size_t)(n0 + wrow) * K + wcol;

  // A operand: lane holds row (m0+li), K chunks {8*hi .. +7} and {8*hi+16 ..}
  const __bf16* ap = A + (size_t)(m0 + li) * K + hi * 8;

  v8f acc[4];
#pragma unroll
  for (int t = 0; t < 4; t++) acc[t] = vzero8();

  // issue one 16B async global->LDS load per lane for chunk kb into buffer bufi
  auto issue = [&](int kb, int bufi) {
    unsigned dst = (unsigned)(size_t)&wbuf[bufi][wrow * 32 + wcol];
    unsigned long long src = (unsigned long long)(size_t)(wsrc0 + kb);
    asm volatile("global_load_async_to_lds_b128 %0, %1, off"
                 :: "v"(dst), "v"(src) : "memory");
  };

  issue(0, 0);
  int cur = 0;
  for (int kb = 0; kb < K; kb += 32, cur ^= 1) {
    if (kb + 32 < K) {
      issue(kb + 32, cur ^ 1);                          // prefetch next chunk
      asm volatile("s_wait_asynccnt 0x1" ::: "memory"); // current chunk done
    } else {
      asm volatile("s_wait_asynccnt 0x0" ::: "memory");
    }
    __syncthreads();   // LDS tile visible to all 8 waves

    v16bf a = ld16(ap + kb, ap + kb + 16);
    __builtin_prefetch(ap + kb + 64, 0, 3);   // global_prefetch_b8
#pragma unroll
    for (int t = 0; t < 4; t++) {
      // B operand from LDS: lane = W row (t*16+li), K chunks {8*hi .. +7} and {8*hi+16 ..}
      const __bf16* wp = &wbuf[cur][(t * 16 + li) * 32 + hi * 8];
      v16bf bm = ld16(wp, wp + 16);
      acc[t] = wmma_bf16(a, bm, acc[t]);
    }
    __syncthreads();   // reads done before this buffer is overwritten
  }

  // D layout: VGPR r -> row (m0 + r + 8*hi), lane -> col (n0 + t*16 + li)
#pragma unroll
  for (int t = 0; t < 4; t++) {
    const int col = n0 + t * 16 + li;
    const float bv = bias[col];
#pragma unroll
    for (int r = 0; r < 8; r++) {
      const int row = m0 + r + 8 * hi;
      float v = acc[t][r] + bv;
      if (QSCALE && col < qcols) v *= qscale;
      const size_t o = (size_t)row * N + col;
      if (RESID) v += resid[o];
      if (WF32) outf[o] = v;
      if (WBF)  outb[o] = (__bf16)v;
    }
  }
}

// ---------------------------------------------------------------------------
// Block-sparse flash attention.  One wave per (b, h, block-row m):
//  - Q: 8 real queries padded to a 16-row A tile (rows 8..15 duplicate 0..7)
//  - loop over 32-key chunks, skip chunks whose mask is entirely zero
//    (mask is shared across b,h so the ballot is wave-uniform)
//  - scores via 2x v_wmma (bf16), online softmax with half-wave reductions
//    (D-tile rows live per 16-lane half), P -> LDS -> A-layout reload,
//    P@V via 2x v_wmma.  Output bf16 [S,B,D].
// ---------------------------------------------------------------------------
__global__ __launch_bounds__(256) void k_attn(const __bf16* __restrict__ qkv,
                                              const unsigned char* __restrict__ maskl,
                                              __bf16* __restrict__ ao) {
  __shared__ __bf16 plds[8][16 * 32];   // 1KB per wave
  const int lane = threadIdx.x & 31;
  const int wave = threadIdx.x >> 5;
  const int li = lane & 15;
  const int hi = lane >> 4;
  const int aoff = hi * 8;    // A-operand K offset
  const int boff = hi * 16;   // unused (B tiles also use aoff)

  const int wid = blockIdx.x * 8 + wave;      // B*NHEAD*MBLK = 8192 waves
  const int b = wid >> 11;
  const int h = (wid >> 8) & 7;
  const int m = wid & 255;

  // Q A-tile (loaded once; q scale already folded in by the qkv GEMM)
  const int sq = m * VECQ + (li & 7);
  const __bf16* qp = qkv + (size_t)(sq * BATCH + b) * QKVW + h * HDIM;
  const v16bf aq = ld16(qp + aoff, qp + aoff + 16);

  float runmax[8], runsum[8];
  v8f O0 = vzero8(), O1 = vzero8();
#pragma unroll
  for (int r = 0; r < 8; r++) { runmax[r] = -3.0e38f; runsum[r] = 0.f; }

  __bf16* pl = &plds[wave][0];

  for (int kc = 0; kc < S_LEN; kc += 32) {
    const int mb = (int)maskl[m * S_LEN + kc + lane];
    if (__ballot(mb != 0) == 0ull) continue;   // uniform skip: sparsity win

    // ---- scores for 2 x 16 keys -----------------------------------------
    const int k0 = kc + li;
    const int k1 = kc + 16 + li;
    const __bf16* kp0 = qkv + (size_t)(k0 * BATCH + b) * QKVW + DMODEL + h * HDIM + aoff;
    const __bf16* kp1 = qkv + (size_t)(k1 * BATCH + b) * QKVW + DMODEL + h * HDIM + aoff;
    v16bf bk0 = ld16(kp0, kp0 + 16);
    v16bf bk1 = ld16(kp1, kp1 + 16);
    v8f s0 = wmma_bf16(aq, bk0, vzero8());
    v8f s1 = wmma_bf16(aq, bk1, vzero8());

    // mask (per key column; broadcast each lane's byte to its column users)
    const int m0b = __shfl(mb, li, 32);
    const int m1b = __shfl(mb, li + 16, 32);
#pragma unroll
    for (int r = 0; r < 8; r++) {
      if (!m0b) s0[r] = -3.0e38f;
      if (!m1b) s1[r] = -3.0e38f;
    }

    // ---- online softmax (rows live per 16-lane half) ---------------------
    float p0[8], p1[8];
#pragma unroll
    for (int r = 0; r < 8; r++) {
      float cm = fmaxf(s0[r], s1[r]);
#pragma unroll
      for (int xm = 8; xm >= 1; xm >>= 1) cm = fmaxf(cm, __shfl_xor(cm, xm, 32));
      const float nm = fmaxf(runmax[r], cm);
      const float corr = __expf(runmax[r] - nm);
      runmax[r] = nm;
      p0[r] = __expf(s0[r] - nm);
      p1[r] = __expf(s1[r] - nm);
      float rs = p0[r] + p1[r];
#pragma unroll
      for (int xm = 8; xm >= 1; xm >>= 1) rs += __shfl_xor(rs, xm, 32);
      runsum[r] = runsum[r] * corr + rs;
      O0[r] *= corr;
      O1[r] *= corr;
    }

    // ---- P (f32, D layout) -> LDS row-major 16x32 -> bf16 A layout -------
#pragma unroll
    for (int r = 0; r < 8; r++) {
      const int prow = r + 8 * hi;
      pl[prow * 32 + li]      = (__bf16)p0[r];
      pl[prow * 32 + 16 + li] = (__bf16)p1[r];
    }
    asm volatile("s_wait_dscnt 0" ::: "memory");   // DS in-order per wave
    v16bf apv = ld16(pl + li * 32 + aoff, pl + li * 32 + aoff + 16);

    // ---- V B-tiles: keys {aoff..+7} and {aoff+16..+7}, col = hd ------------
    v16bf bv0, bv1;
#pragma unroll
    for (int i = 0; i < 16; i++) {
      const int key = kc + aoff + (i & 7) + ((i >> 3) << 4);
      const __bf16* vp = qkv + (size_t)(key * BATCH + b) * QKVW + 2 * DMODEL + h * HDIM;
      bv0[i] = vp[li];
      bv1[i] = vp[16 + li];
    }
    O0 = wmma_bf16(apv, bv0, O0);
    O1 = wmma_bf16(apv, bv1, O1);
  }

  // rows 0..7 (lanes 0..15) are the real queries
  if (hi == 0) {
#pragma unroll
    for (int r = 0; r < 8; r++) {
      const float inv = 1.f / runsum[r];
      const int so = m * VECQ + r;
      __bf16* op = ao + (size_t)(so * BATCH + b) * DMODEL + h * HDIM;
      op[li]      = (__bf16)(O0[r] * inv);
      op[16 + li] = (__bf16)(O1[r] * inv);
    }
  }
}

// ---------------------------------------------------------------------------
// Classifier: out[row, c] = sum_d o[row,d] * wf[c,d] + bf[c];  NC = 2
// One wave per token row.
// ---------------------------------------------------------------------------
__global__ __launch_bounds__(256) void k_cls(const __bf16* __restrict__ o,
                                             const float* __restrict__ wf,
                                             const float* __restrict__ bfv,
                                             float* __restrict__ out) {
  const int lane = threadIdx.x & 31;
  const int row  = blockIdx.x * 8 + (threadIdx.x >> 5);
  const __bf16* op = o + (size_t)row * DMODEL;
#pragma unroll
  for (int c = 0; c < 2; c++) {
    float s = 0.f;
    for (int d = lane; d < DMODEL; d += 32)
      s += (float)op[d] * wf[c * DMODEL + d];
#pragma unroll
    for (int m = 16; m >= 1; m >>= 1) s += __shfl_xor(s, m, 32);
    if (lane == 0) out[row * 2 + c] = s + bfv[c];
  }
}

// ---------------------------------------------------------------------------
// Host orchestration
// ---------------------------------------------------------------------------
extern "C" void kernel_launch(void* const* d_in, const int* in_sizes, int n_in,
                              void* d_out, int out_size, void* d_ws, size_t ws_size,
                              hipStream_t stream) {
  const int*            x     = (const int*)d_in[0];
  const unsigned char*  mask  = (const unsigned char*)d_in[1];
  const float* emb  = (const float*)d_in[2];
  const float* ln1g = (const float*)d_in[3];
  const float* ln1b = (const float*)d_in[4];
  const float* in_w = (const float*)d_in[5];
  const float* in_b = (const float*)d_in[6];
  const float* ow   = (const float*)d_in[7];
  const float* ob   = (const float*)d_in[8];
  const float* ln2g = (const float*)d_in[9];
  const float* ln2b = (const float*)d_in[10];
  const float* w1   = (const float*)d_in[11];
  const float* b1   = (const float*)d_in[12];
  const float* w2   = (const float*)d_in[13];
  const float* b2   = (const float*)d_in[14];
  const float* lnfg = (const float*)d_in[15];
  const float* lnfb = (const float*)d_in[16];
  const float* wf   = (const float*)d_in[17];
  const float* bfv  = (const float*)d_in[18];
  float* out = (float*)d_out;

  // ---- workspace layout (bytes) ----
  char* ws = (char*)d_ws;
  float*  h    = (float*)(ws + 0);            //  8,388,608  h [S,B,D] f32
  __bf16* obf  = (__bf16*)(ws + 8388608);     //  4,194,304  LN output bf16
  __bf16* qkv  = (__bf16*)(ws + 12582912);    // 12,582,912  qkv bf16 (q pre-scaled)
  __bf16* aob  = (__bf16*)(ws + 25165824);    //  4,194,304  attn out bf16
  float*  h2   = (float*)(ws + 29360128);     //  8,388,608  attn residual f32
  __bf16* ff1  = (__bf16*)(ws + 37748736);    // 16,777,216  mlp hidden bf16
  __bf16* winb = (__bf16*)(ws + 54525952);    //  1,572,864
  __bf16* wob  = (__bf16*)(ws + 56098816);    //    524,288
  __bf16* w1b  = (__bf16*)(ws + 56623104);    //  2,097,152
  __bf16* w2b  = (__bf16*)(ws + 58720256);    //  2,097,152  -> total 60,817,408 B
  (void)ws_size; (void)in_sizes; (void)n_in; (void)out_size;

  const float qscale = 0.17677669529663689f;  // 1/sqrt(HD)

  // weights -> bf16 (L2-resident afterwards)
  k_cast_bf16<<<1024, 256, 0, stream>>>(in_w, winb, NLAYER * QKVW * DMODEL);
  k_cast_bf16<<<1024, 256, 0, stream>>>(ow,   wob,  NLAYER * DMODEL * DMODEL);
  k_cast_bf16<<<1024, 256, 0, stream>>>(w1,   w1b,  NLAYER * MLPD * DMODEL);
  k_cast_bf16<<<1024, 256, 0, stream>>>(w2,   w2b,  NLAYER * DMODEL * MLPD);

  // h = emb[x]*sqrt(D) + pos
  k_embed<<<(S_LEN * BATCH * (DMODEL / 2)) / 256, 256, 0, stream>>>(x, emb, h);

  for (int l = 0; l < NLAYER; l++) {
    // LN1 -> bf16
    k_ln<<<ROWS / 8, 256, 0, stream>>>(h, ln1g + l * DMODEL, ln1b + l * DMODEL, obf);
    // qkv = o @ in_w^T + in_b  (q columns pre-scaled by 1/sqrt(HD))
    k_gemm<true, false, false, true><<<dim3(QKVW / 64, ROWS / 128), 256, 0, stream>>>(
        obf, winb + (size_t)l * QKVW * DMODEL, in_b + l * QKVW,
        qkv, nullptr, nullptr, ROWS, QKVW, DMODEL, qscale, DMODEL);
    // block-sparse flash attention
    k_attn<<<(BATCH * NHEAD * MBLK) / 8, 256, 0, stream>>>(
        qkv, mask + (size_t)l * MBLK * S_LEN, aob);
    // h2 = ao @ out_w^T + out_b + h
    k_gemm<false, true, true, false><<<dim3(DMODEL / 64, ROWS / 128), 256, 0, stream>>>(
        aob, wob + (size_t)l * DMODEL * DMODEL, ob + l * DMODEL,
        nullptr, h2, h, ROWS, DMODEL, DMODEL, 1.f, 0);
    // LN2 -> bf16
    k_ln<<<ROWS / 8, 256, 0, stream>>>(h2, ln2g + l * DMODEL, ln2b + l * DMODEL, obf);
    // ff1 = o2 @ w1^T + b1   (no activation, per reference)
    k_gemm<true, false, false, false><<<dim3(MLPD / 64, ROWS / 128), 256, 0, stream>>>(
        obf, w1b + (size_t)l * MLPD * DMODEL, b1 + l * MLPD,
        ff1, nullptr, nullptr, ROWS, MLPD, DMODEL, 1.f, 0);
    // h = ff1 @ w2^T + b2 + h   (residual uses BLOCK INPUT h, per reference)
    k_gemm<false, true, true, false><<<dim3(DMODEL / 64, ROWS / 128), 256, 0, stream>>>(
        ff1, w2b + (size_t)l * DMODEL * MLPD, b2 + l * DMODEL,
        nullptr, h, h, ROWS, DMODEL, MLPD, 1.f, 0);
  }

  // final LN + classifier
  k_ln<<<ROWS / 8, 256, 0, stream>>>(h, lnfg, lnfb, obf);
  k_cls<<<ROWS / 8, 256, 0, stream>>>(obf, wf, bfv, out);
}
